// TransformerBlock_24404004176270
// MI455X (gfx1250) — matrix operand order repaired and checked by reading, not compiled, not run on hardware
//
#include <hip/hip_runtime.h>

#define S_LEN 2048
#define NH    16
#define DH    64
#define DM    1024
#define DMLP  4096
#define PN    35
#define NODES 52

typedef __attribute__((ext_vector_type(16))) __bf16 v16bf;
typedef __attribute__((ext_vector_type(8)))  __bf16 v8bf;
typedef __attribute__((ext_vector_type(8)))  float  v8f;

static __device__ inline v8f v8f_zero() {
  v8f z;
#pragma unroll
  for (int i = 0; i < 8; ++i) z[i] = 0.0f;
  return z;
}

static __device__ inline v8f wmma_bf16(v16bf a, v16bf b, v8f c) {
  // D = A(16x32 bf16) * B(32x16 bf16) + C(16x16 f32)
  return __builtin_amdgcn_wmma_f32_16x16x32_bf16(false, a, false, b, (short)0, c, false, false);
}

// A fragment: rows row0..row0+15 (M), cols k0..k0+31 (K), row-major source, row stride ld.
static __device__ inline v16bf load_a16(const __bf16* base, int ld, int row0, int k0) {
  const int lane = threadIdx.x & 31;
  const int hl = lane >> 4;
  const int m  = lane & 15;
  const __bf16* r = base + (size_t)(row0 + m) * ld + k0;
  v8bf lo = *(const v8bf*)(r + hl * 16);
  v8bf hi = *(const v8bf*)(r + hl * 16 + 8);
  v16bf a;
#pragma unroll
  for (int i = 0; i < 8; ++i) { a[i] = lo[i]; a[i + 8] = hi[i]; }
  return a;
}

// B fragment (32x16, K x N) from B^T stored row-major as [N][K], row stride ld.
static __device__ inline v16bf load_b16(const __bf16* baseT, int ld, int n0, int k0) {
  const int lane = threadIdx.x & 31;
  const int hl = lane >> 4;
  const int m  = lane & 15;
  const __bf16* r = baseT + (size_t)(n0 + m) * ld + k0 + hl * 16;
  v8bf lo = *(const v8bf*)(r);
  v8bf hi = *(const v8bf*)(r + 8);
  v16bf b;
#pragma unroll
  for (int i = 0; i < 8; ++i) { b[i] = lo[i]; b[i + 8] = hi[i]; }
  return b;
}

static __device__ inline float blk_sum(float v) {
  __shared__ float sh[32];
  const int t = threadIdx.x, lane = t & 31, wid = t >> 5;
#pragma unroll
  for (int o = 16; o >= 1; o >>= 1) v += __shfl_xor(v, o, 32);
  __syncthreads();
  if (lane == 0) sh[wid] = v;
  __syncthreads();
  const int nw = ((int)blockDim.x + 31) >> 5;
  float r = (t < nw) ? sh[t] : 0.0f;
  if (wid == 0) {
#pragma unroll
    for (int o = 16; o >= 1; o >>= 1) r += __shfl_xor(r, o, 32);
    if (lane == 0) sh[0] = r;
  }
  __syncthreads();
  float res = sh[0];
  __syncthreads();
  return res;
}

static __device__ inline void ln_write(const float a[4], __bf16* dst, int m0) {
  float s1 = a[0] + a[1] + a[2] + a[3];
  float s2 = a[0]*a[0] + a[1]*a[1] + a[2]*a[2] + a[3]*a[3];
  s1 = blk_sum(s1);
  s2 = blk_sum(s2);
  const float mean = s1 * (1.0f / DM);
  const float var  = s2 * (1.0f / DM) - mean * mean;
  const float inv  = rsqrtf(var + 1e-5f);
#pragma unroll
  for (int i = 0; i < 4; ++i) dst[m0 + i] = (__bf16)((a[i] - mean) * inv);
}

// ---- transpose + f32->bf16 convert: dst[b][c][r] = src[b][r][c]
__global__ void k_tconv(const float* __restrict__ src, __bf16* __restrict__ dst,
                        int B, int R, int C) {
  const int total = B * R * C;
  for (int i = blockIdx.x * blockDim.x + threadIdx.x; i < total;
       i += gridDim.x * blockDim.x) {
    const int b = i / (R * C);
    const int rem = i - b * R * C;
    const int r = rem / C;
    const int c = rem - r * C;
    dst[((size_t)b * C + c) * R + r] = (__bf16)src[i];
  }
}

// ---- masked node sums + LayerNorm for q/k/v streams (one resid pass per (s,h))
__global__ void k_mask_ln_qkv(const float* __restrict__ resid,
                              const float* __restrict__ mql,
                              const float* __restrict__ mkl,
                              const float* __restrict__ mvl,
                              __bf16* __restrict__ nq, __bf16* __restrict__ nk,
                              __bf16* __restrict__ nv) {
  const int s = blockIdx.x >> 4;
  const int h = blockIdx.x & 15;
  const int m0 = threadIdx.x * 4;
  float aq[4] = {0,0,0,0}, ak[4] = {0,0,0,0}, av[4] = {0,0,0,0};
  for (int p = 0; p < PN; ++p) {
    const float4 v = *(const float4*)(resid + ((size_t)s * PN + p) * DM + m0);
    if (mql[p * NH + h] > 0.0f) { aq[0]+=v.x; aq[1]+=v.y; aq[2]+=v.z; aq[3]+=v.w; }
    if (mkl[p * NH + h] > 0.0f) { ak[0]+=v.x; ak[1]+=v.y; ak[2]+=v.z; ak[3]+=v.w; }
    if (mvl[p * NH + h] > 0.0f) { av[0]+=v.x; av[1]+=v.y; av[2]+=v.z; av[3]+=v.w; }
  }
  __bf16* base = nq + ((size_t)s * NH + h) * DM;
  ln_write(aq, base, m0);
  base = nk + ((size_t)s * NH + h) * DM;
  ln_write(ak, base, m0);
  base = nv + ((size_t)s * NH + h) * DM;
  ln_write(av, base, m0);
}

// ---- QKV projection (WMMA) + bias + rotary; q scaled by 1/sqrt(DH)
__global__ void k_qkv_proj(const __bf16* __restrict__ nq, const __bf16* __restrict__ nk,
                           const __bf16* __restrict__ nv,
                           const __bf16* __restrict__ wqt, const __bf16* __restrict__ wkt,
                           const __bf16* __restrict__ wvt,
                           const float* __restrict__ bQ, const float* __restrict__ bK,
                           const float* __restrict__ bV,
                           __bf16* __restrict__ qb, __bf16* __restrict__ kb,
                           __bf16* __restrict__ vt) {
  const int qt = blockIdx.x;
  const int h = blockIdx.y & 15;
  const int stream = blockIdx.y >> 4;
  const int lane = threadIdx.x & 31;
  const int hl = lane >> 4, m = lane & 15;
  const __bf16* nsrc = (stream == 0) ? nq : (stream == 1) ? nk : nv;
  const __bf16* wt   = (stream == 0) ? wqt : (stream == 1) ? wkt : wvt;
  const float* bias  = ((stream == 0) ? bQ : (stream == 1) ? bK : bV) + h * DH;
  const __bf16* abase = nsrc + (size_t)h * DM;     // row s stride NH*DM
  const __bf16* bbase = wt + (size_t)h * DH * DM;  // [d][m]
  v8f acc[4];
#pragma unroll
  for (int n0 = 0; n0 < 4; ++n0) acc[n0] = v8f_zero();
  for (int kk = 0; kk < DM; kk += 32) {
    v16bf a = load_a16(abase, NH * DM, qt * 16, kk);
#pragma unroll
    for (int n0 = 0; n0 < 4; ++n0)
      acc[n0] = wmma_bf16(a, load_b16(bbase, DM, n0 * 16, kk), acc[n0]);
  }
  if (stream == 2) {
#pragma unroll
    for (int n0 = 0; n0 < 4; ++n0)
#pragma unroll
      for (int r = 0; r < 8; ++r) {
        const int d = n0 * 16 + m;
        const int srow = qt * 16 + hl * 8 + r;
        vt[((size_t)h * DH + d) * S_LEN + srow] = (__bf16)(acc[n0][r] + bias[d]);
      }
  } else {
    __bf16* dst = (stream == 0) ? qb : kb;
    const float sc = (stream == 0) ? 0.125f : 1.0f;  // 1/sqrt(64)
#pragma unroll
    for (int n0 = 0; n0 < 2; ++n0)
#pragma unroll
      for (int r = 0; r < 8; ++r) {
        const int d = n0 * 16 + m;                      // 0..31
        const int srow = qt * 16 + hl * 8 + r;
        const float ang = (float)srow * powf(10000.0f, -(float)d * (1.0f / 32.0f));
        const float sn = sinf(ang), cs = cosf(ang);
        const float x = acc[n0][r] + bias[d];
        const float y = acc[n0 + 2][r] + bias[d + 32];
        dst[((size_t)h * S_LEN + srow) * DH + d]      = (__bf16)((x * cs - y * sn) * sc);
        dst[((size_t)h * S_LEN + srow) * DH + d + 32] = (__bf16)((y * cs + x * sn) * sc);
      }
  }
}

// ---- causal flash attention: S^T = K q^T, online softmax, Z^T = V^T P^T
__global__ void k_attn(const __bf16* __restrict__ qb, const __bf16* __restrict__ kb,
                       const __bf16* __restrict__ vt, __bf16* __restrict__ zb) {
  const int qt = blockIdx.x;
  const int h = blockIdx.y;
  const int lane = threadIdx.x & 31;
  const int hl = lane >> 4, m = lane & 15;
  const __bf16* qh = qb + (size_t)h * S_LEN * DH;
  const __bf16* kh = kb + (size_t)h * S_LEN * DH;
  const __bf16* vh = vt + (size_t)h * DH * S_LEN;

  const v16bf bq0 = load_b16(qh, DH, qt * 16, 0);   // q^T cols d 0..31
  const v16bf bq1 = load_b16(qh, DH, qt * 16, 32);  // d 32..63

  v8f zacc[4];
#pragma unroll
  for (int t = 0; t < 4; ++t) zacc[t] = v8f_zero();
  float mrun = -1e30f, lrun = 0.0f;
  const int qcol = qt * 16 + m;
  const int kend = qt * 16 + 16;

  for (int k0 = 0; k0 < kend; k0 += 32) {
    v8f sa = wmma_bf16(load_a16(kh, DH, k0, 0), bq0, v8f_zero());
    sa = wmma_bf16(load_a16(kh, DH, k0, 32), bq1, sa);
    v8f sb = wmma_bf16(load_a16(kh, DH, k0 + 16, 0), bq0, v8f_zero());
    sb = wmma_bf16(load_a16(kh, DH, k0 + 16, 32), bq1, sb);

    float pa[8], pb[8];
    float tmax = -1e30f;
#pragma unroll
    for (int r = 0; r < 8; ++r) {
      const int ka = k0 + hl * 8 + r;
      const int kc = k0 + 16 + hl * 8 + r;
      pa[r] = (ka <= qcol) ? sa[r] : -1e30f;
      pb[r] = (kc <= qcol) ? sb[r] : -1e30f;
      tmax = fmaxf(tmax, fmaxf(pa[r], pb[r]));
    }
    tmax = fmaxf(tmax, __shfl_xor(tmax, 16, 32));
    const float mnew = fmaxf(mrun, tmax);
    const float fs = __expf(mrun - mnew);
    float psum = 0.0f;
#pragma unroll
    for (int r = 0; r < 8; ++r) {
      pa[r] = __expf(pa[r] - mnew);
      pb[r] = __expf(pb[r] - mnew);
      psum += pa[r] + pb[r];
    }
    psum += __shfl_xor(psum, 16, 32);
    lrun = lrun * fs + psum;
    mrun = mnew;
#pragma unroll
    for (int t = 0; t < 4; ++t)
#pragma unroll
      for (int r = 0; r < 8; ++r) zacc[t][r] *= fs;

    float oa[8], ob[8];
#pragma unroll
    for (int r = 0; r < 8; ++r) {
      oa[r] = __shfl_xor(pa[r], 16, 32);
      ob[r] = __shfl_xor(pb[r], 16, 32);
    }
    v16bf pfrag;  // B-frag of stacked P^T (32 k-rows x 16 q-cols)
#pragma unroll
    for (int e = 0; e < 8; ++e) {
      pfrag[e]     = (__bf16)(hl ? ob[e] : pa[e]);
      pfrag[e + 8] = (__bf16)(hl ? pb[e] : oa[e]);
    }
#pragma unroll
    for (int t = 0; t < 4; ++t)
      zacc[t] = wmma_bf16(load_a16(vh, S_LEN, t * 16, k0), pfrag, zacc[t]);
  }
  const float inv_l = 1.0f / lrun;
#pragma unroll
  for (int t = 0; t < 4; ++t)
#pragma unroll
    for (int r = 0; r < 8; ++r) {
      const int d = t * 16 + hl * 8 + r;
      zb[((size_t)h * S_LEN + qt * 16 + m) * DH + d] = (__bf16)(zacc[t][r] * inv_l);
    }
}

// ---- per-head output projection: writes attn node (35+h) directly into d_out
__global__ void k_attn_out(const __bf16* __restrict__ zb, const __bf16* __restrict__ wot,
                           const float* __restrict__ bO, float* __restrict__ out) {
  const int qt = blockIdx.x;
  const int cb = blockIdx.y;
  const int h = blockIdx.z;
  const int lane = threadIdx.x & 31;
  const int hl = lane >> 4, m = lane & 15;
  const __bf16* zh = zb + (size_t)h * S_LEN * DH;
  const __bf16* wh = wot + (size_t)h * DM * DH;  // [m][d]
  v8f acc[4];
#pragma unroll
  for (int n0 = 0; n0 < 4; ++n0) acc[n0] = v8f_zero();
  for (int kk = 0; kk < DH; kk += 32) {
    v16bf a = load_a16(zh, DH, qt * 16, kk);
#pragma unroll
    for (int n0 = 0; n0 < 4; ++n0)
      acc[n0] = wmma_bf16(a, load_b16(wh, DH, cb * 64 + n0 * 16, kk), acc[n0]);
  }
#pragma unroll
  for (int n0 = 0; n0 < 4; ++n0)
#pragma unroll
    for (int r = 0; r < 8; ++r) {
      const int col = cb * 64 + n0 * 16 + m;
      const int srow = qt * 16 + hl * 8 + r;
      out[((size_t)srow * NODES + PN + h) * DM + col] =
          acc[n0][r] + bO[col] * (1.0f / NH);
    }
}

// ---- copy resid nodes, masked MLP reduction over 51 nodes, LayerNorm
__global__ void k_resid_mlp_ln(const float* __restrict__ resid,
                               const float* __restrict__ mmlp,
                               float* __restrict__ out, __bf16* __restrict__ nm) {
  const int s = blockIdx.x;
  const int m0 = threadIdx.x * 4;
  float acc[4] = {0,0,0,0};
  for (int n = 0; n < PN; ++n) {
    const float4 v = *(const float4*)(resid + ((size_t)s * PN + n) * DM + m0);
    *(float4*)(out + ((size_t)s * NODES + n) * DM + m0) = v;
    if (mmlp[n] > 0.0f) { acc[0]+=v.x; acc[1]+=v.y; acc[2]+=v.z; acc[3]+=v.w; }
  }
  for (int hh = 0; hh < NH; ++hh) {
    if (mmlp[PN + hh] > 0.0f) {
      const float4 v = *(const float4*)(out + ((size_t)s * NODES + PN + hh) * DM + m0);
      acc[0]+=v.x; acc[1]+=v.y; acc[2]+=v.z; acc[3]+=v.w;
    }
  }
  ln_write(acc, nm + (size_t)s * DM, m0);
}

// ---- MLP GEMM 1 + gelu(tanh)
__global__ void k_mlp1(const __bf16* __restrict__ nm, const __bf16* __restrict__ wint,
                       const float* __restrict__ bin, __bf16* __restrict__ hb) {
  const int qt = blockIdx.x;
  const int jb = blockIdx.y;
  const int lane = threadIdx.x & 31;
  const int hl = lane >> 4, m = lane & 15;
  v8f acc[4];
#pragma unroll
  for (int n0 = 0; n0 < 4; ++n0) acc[n0] = v8f_zero();
  for (int kk = 0; kk < DM; kk += 32) {
    v16bf a = load_a16(nm, DM, qt * 16, kk);
    if (kk + 64 < DM)
      __builtin_prefetch((const void*)(nm + (size_t)(qt * 16 + m) * DM + kk + 64), 0, 0);
#pragma unroll
    for (int n0 = 0; n0 < 4; ++n0)
      acc[n0] = wmma_bf16(a, load_b16(wint, DM, jb * 64 + n0 * 16, kk), acc[n0]);
  }
#pragma unroll
  for (int n0 = 0; n0 < 4; ++n0)
#pragma unroll
    for (int r = 0; r < 8; ++r) {
      const int j = jb * 64 + n0 * 16 + m;
      const int srow = qt * 16 + hl * 8 + r;
      const float x = acc[n0][r] + bin[j];
      const float g = 0.5f * x * (1.0f + tanhf(0.7978845608f * (x + 0.044715f * x * x * x)));
      hb[(size_t)srow * DMLP + j] = (__bf16)g;
    }
}

// ---- MLP GEMM 2: writes node 51 into d_out
__global__ void k_mlp2(const __bf16* __restrict__ hb, const __bf16* __restrict__ woutt,
                       const float* __restrict__ bout, float* __restrict__ out) {
  const int qt = blockIdx.x;
  const int cb = blockIdx.y;
  const int lane = threadIdx.x & 31;
  const int hl = lane >> 4, m = lane & 15;
  v8f acc[4];
#pragma unroll
  for (int n0 = 0; n0 < 4; ++n0) acc[n0] = v8f_zero();
  for (int kk = 0; kk < DMLP; kk += 32) {
    v16bf a = load_a16(hb, DMLP, qt * 16, kk);
    if (kk + 64 < DMLP)
      __builtin_prefetch((const void*)(hb + (size_t)(qt * 16 + m) * DMLP + kk + 64), 0, 0);
#pragma unroll
    for (int n0 = 0; n0 < 4; ++n0)
      acc[n0] = wmma_bf16(a, load_b16(woutt, DMLP, cb * 64 + n0 * 16, kk), acc[n0]);
  }
#pragma unroll
  for (int n0 = 0; n0 < 4; ++n0)
#pragma unroll
    for (int r = 0; r < 8; ++r) {
      const int col = cb * 64 + n0 * 16 + m;
      const int srow = qt * 16 + hl * 8 + r;
      out[((size_t)srow * NODES + (NODES - 1)) * DM + col] = acc[n0][r] + bout[col];
    }
}

extern "C" void kernel_launch(void* const* d_in, const int* in_sizes, int n_in,
                              void* d_out, int out_size, void* d_ws, size_t ws_size,
                              hipStream_t stream) {
  (void)in_sizes; (void)n_in; (void)out_size; (void)ws_size;
  const float* resid = (const float*)d_in[0];
  const float* WQ   = (const float*)d_in[1];
  const float* bQ   = (const float*)d_in[2];
  const float* WK   = (const float*)d_in[3];
  const float* bK   = (const float*)d_in[4];
  const float* WV   = (const float*)d_in[5];
  const float* bV   = (const float*)d_in[6];
  const float* WO   = (const float*)d_in[7];
  const float* bO   = (const float*)d_in[8];
  const float* mql  = (const float*)d_in[9];
  const float* mkl  = (const float*)d_in[10];
  const float* mvl  = (const float*)d_in[11];
  const float* mmlp = (const float*)d_in[12];
  const float* Win  = (const float*)d_in[13];
  const float* bin  = (const float*)d_in[14];
  const float* Wout = (const float*)d_in[15];
  const float* bout = (const float*)d_in[16];
  float* out = (float*)d_out;

  char* ws = (char*)d_ws;
  size_t off = 0;
  auto take = [&](size_t bytes) -> char* {
    char* p = ws + off;
    off += (bytes + 255) & ~(size_t)255;
    return p;
  };
  __bf16* nq    = (__bf16*)take((size_t)S_LEN * NH * DM * 2);
  __bf16* nk    = (__bf16*)take((size_t)S_LEN * NH * DM * 2);
  __bf16* nv    = (__bf16*)take((size_t)S_LEN * NH * DM * 2);
  __bf16* wqt   = (__bf16*)take((size_t)NH * DH * DM * 2);
  __bf16* wkt   = (__bf16*)take((size_t)NH * DH * DM * 2);
  __bf16* wvt   = (__bf16*)take((size_t)NH * DH * DM * 2);
  __bf16* wot   = (__bf16*)take((size_t)NH * DM * DH * 2);
  __bf16* qb    = (__bf16*)take((size_t)NH * S_LEN * DH * 2);
  __bf16* kbuf  = (__bf16*)take((size_t)NH * S_LEN * DH * 2);
  __bf16* vtb   = (__bf16*)take((size_t)NH * DH * S_LEN * 2);
  __bf16* zb    = (__bf16*)take((size_t)NH * S_LEN * DH * 2);
  __bf16* nm    = (__bf16*)take((size_t)S_LEN * DM * 2);
  __bf16* wint  = (__bf16*)take((size_t)DMLP * DM * 2);
  __bf16* woutt = (__bf16*)take((size_t)DM * DMLP * 2);
  __bf16* hb    = (__bf16*)take((size_t)S_LEN * DMLP * 2);

  // weight conversions (transpose to [N][K] bf16)
  k_tconv<<<dim3(1024), dim3(256), 0, stream>>>(WQ, wqt, NH, DM, DH);
  k_tconv<<<dim3(1024), dim3(256), 0, stream>>>(WK, wkt, NH, DM, DH);
  k_tconv<<<dim3(1024), dim3(256), 0, stream>>>(WV, wvt, NH, DM, DH);
  k_tconv<<<dim3(1024), dim3(256), 0, stream>>>(WO, wot, NH, DH, DM);
  k_tconv<<<dim3(4096), dim3(256), 0, stream>>>(Win, wint, 1, DM, DMLP);
  k_tconv<<<dim3(4096), dim3(256), 0, stream>>>(Wout, woutt, 1, DMLP, DM);

  k_mask_ln_qkv<<<dim3(S_LEN * NH), dim3(256), 0, stream>>>(resid, mql, mkl, mvl,
                                                            nq, nk, nv);
  k_qkv_proj<<<dim3(S_LEN / 16, 3 * NH), dim3(32), 0, stream>>>(
      nq, nk, nv, wqt, wkt, wvt, bQ, bK, bV, qb, kbuf, vtb);
  k_attn<<<dim3(S_LEN / 16, NH), dim3(32), 0, stream>>>(qb, kbuf, vtb, zb);
  k_attn_out<<<dim3(S_LEN / 16, DM / 64, NH), dim3(32), 0, stream>>>(zb, wot, bO, out);
  k_resid_mlp_ln<<<dim3(S_LEN), dim3(256), 0, stream>>>(resid, mmlp, out, nm);
  k_mlp1<<<dim3(S_LEN / 16, DMLP / 64), dim3(32), 0, stream>>>(nm, wint, bin, hb);
  k_mlp2<<<dim3(S_LEN / 16, DM / 64), dim3(32), 0, stream>>>(hb, woutt, bout, out);
}
